// SparseConv_45844480917742
// MI455X (gfx1250) — matrix operand order repaired and checked by reading, not compiled, not run on hardware
//
#include <hip/hip_runtime.h>
#include <hip/hip_bf16.h>

// ---------------------------------------------------------------------------
// Sparse conv (output-stationary) for MI455X / gfx1250, wave32 + WMMA bf16.
//   out[n, co] = sum_k sum_ci mask[n,k] * features[kmap[n,k], ci] * W[k, ci, co]
//
// Fast path (ws_size permitting):
//   prep kernel : W f32 [27][cin][cout]  ->  d_ws bf16 [27][cout][cin]
//   main kernel : 256 threads / 8 waves per 128x64 output tile,
//                 double-buffered LDS A (gather+mask+cvt), B fragments
//                 loaded directly from L2-resident pretransposed weights,
//                 software-pipelined gathers, ONE barrier per k-offset.
// Fallback path: self-contained kernel with in-LDS weight transpose.
// ---------------------------------------------------------------------------

#define KVOL       27
#define CIN        64
#define COUT       64
#define TILE_M     128
#define LDS_STRIDE 72   // bf16 elements per A row (144 B, multiple of 16 B)

typedef __attribute__((ext_vector_type(16))) __bf16 v16bf;
typedef __attribute__((ext_vector_type(8)))  float  v8f;

union FragBF16 {
    struct { uint4 lo, hi; } q;   // two 16-byte chunks -> elements 0..7 / 8..15
    v16bf v;
};

static __device__ __forceinline__ unsigned int pack2_bf16(float a, float b) {
    // f32 pair -> packed bf16x2 (RNE); header picks HW cvt_pk path if available
    union { __hip_bfloat162 h; unsigned int u; } c;
    c.h = __float22bfloat162_rn(make_float2(a, b));
    return c.u;
}

static __device__ __forceinline__ unsigned short f2bf(float a) {
    unsigned int ua = __float_as_uint(a);
    return (unsigned short)((ua + 0x7FFFu + ((ua >> 16) & 1u)) >> 16);
}

static __device__ __forceinline__ uint4 pack8_bf16(const float4& x, const float4& y) {
    uint4 p;
    p.x = pack2_bf16(x.x, x.y);
    p.y = pack2_bf16(x.z, x.w);
    p.z = pack2_bf16(y.x, y.y);
    p.w = pack2_bf16(y.z, y.w);
    return p;
}

// ===========================================================================
// Prep: transpose+convert weights once per launch. d_ws[k][cout][cin] (bf16).
// ===========================================================================
__global__ __launch_bounds__(256) void prep_weights_bf16(
    const float* __restrict__ kernelW, unsigned short* __restrict__ wbf)
{
    const int idx = blockIdx.x * 256 + threadIdx.x;
    if (idx < KVOL * CIN * COUT) {
        const int k    = idx / (CIN * COUT);
        const int r    = idx % (CIN * COUT);
        const int cout = r / CIN;
        const int cin  = r % CIN;
        wbf[idx] = f2bf(kernelW[(size_t)k * CIN * COUT + (size_t)cin * COUT + cout]);
    }
}

// ===========================================================================
// Fast path: pretransposed weights, double-buffered A, 1 barrier / k.
// ===========================================================================
__global__ __launch_bounds__(256) void sparse_conv_wmma_fast(
    const float* __restrict__ features,          // [N, 64] f32
    const unsigned short* __restrict__ wbf,      // [27][cout=64][cin=64] bf16
    const long long* __restrict__ kmap,          // [N, 27] i64
    const int* __restrict__ kmask,               // [N, 27] i32
    float* __restrict__ out,                     // [N, 64] f32
    int n_pts)
{
    __shared__ __align__(16) unsigned short As[2][TILE_M * LDS_STRIDE]; // 36 KB

    const int t    = threadIdx.x;
    const int wave = t >> 5;
    const int lane = t & 31;

    // gather-stage mapping: 2 threads per row, 32 channels each
    const int grow   = t >> 1;
    const int cstart = (t & 1) * 32;
    const long long gabs = (long long)blockIdx.x * TILE_M + grow;
    const bool grow_ok = (gabs < (long long)n_pts);

    // WMMA fragment lane mapping (ISA 7.12.2)
    const int mbase = wave * 16;
    const int am    = lane & 15;
    const int akoff = (lane >> 4) * 16;   // A K-split: {0..15} / {16..31} contiguous
    const int bn    = lane & 15;
    const int bkoff = (lane >> 4) * 16;   // B K-split: {0..15} / {16..31} contiguous

    v8f acc0 = {}, acc1 = {}, acc2 = {}, acc3 = {};

    // ---- pipeline prologue: gather k=0, prefetch indices for k=1 ----------
    {
        long long idx0 = 0; float msk0 = 0.0f;
        if (grow_ok) {
            idx0 = kmap [gabs * KVOL + 0];
            msk0 = (float)kmask[gabs * KVOL + 0];
        }
        unsigned short* adst = &As[0][grow * LDS_STRIDE + cstart];
        if (msk0 != 0.0f) {
            const float* src = features + idx0 * (long long)CIN + cstart;
            #pragma unroll
            for (int g = 0; g < 4; ++g) {
                float4 x = *(const float4*)(src + g * 8);
                float4 y = *(const float4*)(src + g * 8 + 4);
                *(uint4*)(adst + g * 8) = pack8_bf16(x, y);
            }
        } else {
            const uint4 z = {0u, 0u, 0u, 0u};
            #pragma unroll
            for (int g = 0; g < 4; ++g) *(uint4*)(adst + g * 8) = z;
        }
    }
    long long idx_nxt = 0; float msk_nxt = 0.0f;
    if (grow_ok) {
        idx_nxt = kmap [gabs * KVOL + 1];
        msk_nxt = (float)kmask[gabs * KVOL + 1];
    }
    __syncthreads();

    for (int k = 0; k < KVOL; ++k) {
        const int buf = k & 1;
        const bool do_next = (k + 1) < KVOL;
        const bool have    = do_next && (msk_nxt != 0.0f);

        // (1) issue k+1 feature gathers early: latency hides under WMMA below
        float4 fx0, fy0, fx1, fy1, fx2, fy2, fx3, fy3;
        if (have) {
            const float* src = features + idx_nxt * (long long)CIN + cstart;
            fx0 = *(const float4*)(src +  0); fy0 = *(const float4*)(src +  4);
            fx1 = *(const float4*)(src +  8); fy1 = *(const float4*)(src + 12);
            fx2 = *(const float4*)(src + 16); fy2 = *(const float4*)(src + 20);
            fx3 = *(const float4*)(src + 24); fy3 = *(const float4*)(src + 28);
        }

        // (2) prefetch indices for k+2
        long long idx_n2 = 0; float msk_n2 = 0.0f;
        if (grow_ok && (k + 2) < KVOL) {
            idx_n2 = kmap [gabs * KVOL + (k + 2)];
            msk_n2 = (float)kmask[gabs * KVOL + (k + 2)];
        }

        // (3) WMMA on buffer `buf`; B fragments straight from L2-hot weights
        const unsigned short* wk = wbf + (size_t)k * (CIN * COUT);
        #pragma unroll
        for (int kk = 0; kk < CIN; kk += 32) {
            FragBF16 af;
            {
                const unsigned short* ap =
                    &As[buf][(mbase + am) * LDS_STRIDE + kk + akoff];
                af.q.lo = *(const uint4*)(ap);       // K = kk+akoff    .. +7
                af.q.hi = *(const uint4*)(ap + 8);   // K = kk+akoff+8  .. +15
            }
            FragBF16 b0, b1, b2, b3;
            {
                const unsigned short* p0 = wk + ( 0 + bn) * CIN + kk + bkoff;
                const unsigned short* p1 = wk + (16 + bn) * CIN + kk + bkoff;
                const unsigned short* p2 = wk + (32 + bn) * CIN + kk + bkoff;
                const unsigned short* p3 = wk + (48 + bn) * CIN + kk + bkoff;
                b0.q.lo = *(const uint4*)(p0); b0.q.hi = *(const uint4*)(p0 + 8);
                b1.q.lo = *(const uint4*)(p1); b1.q.hi = *(const uint4*)(p1 + 8);
                b2.q.lo = *(const uint4*)(p2); b2.q.hi = *(const uint4*)(p2 + 8);
                b3.q.lo = *(const uint4*)(p3); b3.q.hi = *(const uint4*)(p3 + 8);
            }
            acc0 = __builtin_amdgcn_wmma_f32_16x16x32_bf16(
                       false, af.v, false, b0.v, (short)0, acc0, false, false);
            acc1 = __builtin_amdgcn_wmma_f32_16x16x32_bf16(
                       false, af.v, false, b1.v, (short)0, acc1, false, false);
            acc2 = __builtin_amdgcn_wmma_f32_16x16x32_bf16(
                       false, af.v, false, b2.v, (short)0, acc2, false, false);
            acc3 = __builtin_amdgcn_wmma_f32_16x16x32_bf16(
                       false, af.v, false, b3.v, (short)0, acc3, false, false);
        }

        // (4) convert + store k+1 gathers into the other buffer
        if (do_next) {
            unsigned short* adst = &As[buf ^ 1][grow * LDS_STRIDE + cstart];
            if (have) {
                *(uint4*)(adst +  0) = pack8_bf16(fx0, fy0);
                *(uint4*)(adst +  8) = pack8_bf16(fx1, fy1);
                *(uint4*)(adst + 16) = pack8_bf16(fx2, fy2);
                *(uint4*)(adst + 24) = pack8_bf16(fx3, fy3);
            } else {
                const uint4 z = {0u, 0u, 0u, 0u};
                *(uint4*)(adst +  0) = z; *(uint4*)(adst +  8) = z;
                *(uint4*)(adst + 16) = z; *(uint4*)(adst + 24) = z;
            }
        }
        idx_nxt = idx_n2; msk_nxt = msk_n2;

        __syncthreads();   // k+1 A visible; everyone done reading buf
    }

    // ---- store C/D (16x16 f32 layout), tail-guarded ------------------------
    const int crow0 = (lane < 16) ? 0 : 8;
    const int ccol  = lane & 15;
    const long long obase = (long long)blockIdx.x * TILE_M + mbase + crow0;
    #pragma unroll
    for (int r = 0; r < 8; ++r) {
        const long long orow = obase + r;
        if (orow < (long long)n_pts) {
            float* op = out + orow * COUT + ccol;
            op[ 0] = acc0[r];
            op[16] = acc1[r];
            op[32] = acc2[r];
            op[48] = acc3[r];
        }
    }
}

// ===========================================================================
// Fallback path (no workspace): in-LDS weight transpose each k, 2 barriers/k.
// ===========================================================================
__global__ __launch_bounds__(256) void sparse_conv_wmma_bf16(
    const float* __restrict__ features,
    const float* __restrict__ kernelW,
    const long long* __restrict__ kmap,
    const int* __restrict__ kmask,
    float* __restrict__ out,
    int n_pts)
{
    __shared__ __align__(16) unsigned short As[TILE_M * LDS_STRIDE];
    __shared__ __align__(16) unsigned short Bs[COUT  * LDS_STRIDE];

    const int t    = threadIdx.x;
    const int wave = t >> 5;
    const int lane = t & 31;

    const int grow   = t >> 1;
    const int cstart = (t & 1) * 32;
    const long long gabs = (long long)blockIdx.x * TILE_M + grow;
    const bool grow_ok = (gabs < (long long)n_pts);

    const int wcin   = t >> 2;
    const int wcout0 = (t & 3) * 16;

    const int mbase = wave * 16;
    const int am    = lane & 15;
    const int akoff = (lane >> 4) * 16;
    const int bn    = lane & 15;
    const int bkoff = (lane >> 4) * 16;

    v8f acc0 = {}, acc1 = {}, acc2 = {}, acc3 = {};

    for (int k = 0; k < KVOL; ++k) {
        {
            float msk = 0.0f; long long idx = 0;
            if (grow_ok) {
                idx = kmap [gabs * KVOL + k];
                msk = (float)kmask[gabs * KVOL + k];
            }
            unsigned short* adst = &As[grow * LDS_STRIDE + cstart];
            if (grow_ok && msk != 0.0f) {
                const float* src = features + idx * (long long)CIN + cstart;
                #pragma unroll
                for (int g = 0; g < 4; ++g) {
                    float4 x = *(const float4*)(src + g * 8);
                    float4 y = *(const float4*)(src + g * 8 + 4);
                    *(uint4*)(adst + g * 8) = pack8_bf16(x, y);
                }
            } else {
                const uint4 z = {0u, 0u, 0u, 0u};
                #pragma unroll
                for (int g = 0; g < 4; ++g) *(uint4*)(adst + g * 8) = z;
            }
        }
        {
            const float* wsrc = kernelW + (size_t)k * (CIN * COUT)
                                        + (size_t)wcin * COUT + wcout0;
            #pragma unroll
            for (int j4 = 0; j4 < 4; ++j4) {
                float4 w = *(const float4*)(wsrc + j4 * 4);
                const int cb = wcout0 + j4 * 4;
                Bs[(cb + 0) * LDS_STRIDE + wcin] = f2bf(w.x);
                Bs[(cb + 1) * LDS_STRIDE + wcin] = f2bf(w.y);
                Bs[(cb + 2) * LDS_STRIDE + wcin] = f2bf(w.z);
                Bs[(cb + 3) * LDS_STRIDE + wcin] = f2bf(w.w);
            }
        }
        __syncthreads();

        #pragma unroll
        for (int kk = 0; kk < CIN; kk += 32) {
            FragBF16 af;
            {
                const unsigned short* ap = &As[(mbase + am) * LDS_STRIDE + kk + akoff];
                af.q.lo = *(const uint4*)(ap);
                af.q.hi = *(const uint4*)(ap + 8);
            }
            FragBF16 b0, b1, b2, b3;
            {
                const unsigned short* p0 = &Bs[( 0 + bn) * LDS_STRIDE + kk + bkoff];
                const unsigned short* p1 = &Bs[(16 + bn) * LDS_STRIDE + kk + bkoff];
                const unsigned short* p2 = &Bs[(32 + bn) * LDS_STRIDE + kk + bkoff];
                const unsigned short* p3 = &Bs[(48 + bn) * LDS_STRIDE + kk + bkoff];
                b0.q.lo = *(const uint4*)(p0); b0.q.hi = *(const uint4*)(p0 + 8);
                b1.q.lo = *(const uint4*)(p1); b1.q.hi = *(const uint4*)(p1 + 8);
                b2.q.lo = *(const uint4*)(p2); b2.q.hi = *(const uint4*)(p2 + 8);
                b3.q.lo = *(const uint4*)(p3); b3.q.hi = *(const uint4*)(p3 + 8);
            }
            acc0 = __builtin_amdgcn_wmma_f32_16x16x32_bf16(
                       false, af.v, false, b0.v, (short)0, acc0, false, false);
            acc1 = __builtin_amdgcn_wmma_f32_16x16x32_bf16(
                       false, af.v, false, b1.v, (short)0, acc1, false, false);
            acc2 = __builtin_amdgcn_wmma_f32_16x16x32_bf16(
                       false, af.v, false, b2.v, (short)0, acc2, false, false);
            acc3 = __builtin_amdgcn_wmma_f32_16x16x32_bf16(
                       false, af.v, false, b3.v, (short)0, acc3, false, false);
        }
        __syncthreads();
    }

    const int crow0 = (lane < 16) ? 0 : 8;
    const int ccol  = lane & 15;
    const long long obase = (long long)blockIdx.x * TILE_M + mbase + crow0;
    #pragma unroll
    for (int r = 0; r < 8; ++r) {
        const long long orow = obase + r;
        if (orow < (long long)n_pts) {
            float* op = out + orow * COUT + ccol;
            op[ 0] = acc0[r];
            op[16] = acc1[r];
            op[32] = acc2[r];
            op[48] = acc3[r];
        }
    }
}

extern "C" void kernel_launch(void* const* d_in, const int* in_sizes, int n_in,
                              void* d_out, int out_size, void* d_ws, size_t ws_size,
                              hipStream_t stream) {
    const float*     features = (const float*)d_in[0];     // [N, 64] f32
    const float*     kernelW  = (const float*)d_in[1];     // [27, 64, 64] f32
    const long long* kmap     = (const long long*)d_in[2]; // [N, 27] i64
    const int*       kmask    = (const int*)d_in[3];       // [N, 27] i32
    float*           out      = (float*)d_out;             // [N, 64] f32

    const int n_pts = in_sizes[0] / CIN;
    const int grid  = (n_pts + TILE_M - 1) / TILE_M;

    const size_t ws_need = (size_t)KVOL * CIN * COUT * sizeof(unsigned short);
    if (ws_size >= ws_need) {
        unsigned short* wbf = (unsigned short*)d_ws;
        prep_weights_bf16<<<(KVOL * CIN * COUT + 255) / 256, 256, 0, stream>>>(
            kernelW, wbf);
        sparse_conv_wmma_fast<<<grid, 256, 0, stream>>>(
            features, wbf, kmap, kmask, out, n_pts);
    } else {
        sparse_conv_wmma_bf16<<<grid, 256, 0, stream>>>(
            features, kernelW, kmap, kmask, out, n_pts);
    }
    (void)n_in; (void)out_size;
}
